// aformer_34119220200098
// MI455X (gfx1250) — matrix operand order repaired and checked by reading, not compiled, not run on hardware
//
#include <hip/hip_runtime.h>
#include <hip/hip_bf16.h>
#include <math.h>

typedef __attribute__((ext_vector_type(16))) __bf16 v16bf;
typedef __attribute__((ext_vector_type(8)))  float  v8f;

// ---------------------------------------------------------------------------
// helpers
// ---------------------------------------------------------------------------
__device__ __forceinline__ unsigned short f2bf(float f) {
    unsigned int u = __builtin_bit_cast(unsigned int, f);
    u += 0x7FFFu + ((u >> 16) & 1u);          // round-to-nearest-even
    return (unsigned short)(u >> 16);
}
__device__ __forceinline__ __bf16 us2bf(unsigned short u) {
    return __builtin_bit_cast(__bf16, u);
}
__device__ __forceinline__ float gelu_f(float x) {
    return 0.5f * x * (1.0f + erff(x * 0.70710678118654752f));
}

// ---------------------------------------------------------------------------
// Tiled bf16 WMMA GEMM:  C[M,N] = A * B (+ bias), f32 accumulate.
//   A(m,k) = A[m*sAm + k*sAk]   (bf16 bits in ushort), ACONT => sAk==1
//   B(k,n) = B[k*sBk + n*sBn]
//
// Block = 256 threads (8 waves) computes a 128(M) x 64(N) macro-tile.
// Waves: wm = wid&3 selects 32-row band, wn = wid>>2 selects 32-col band;
// each wave computes 2x2 WMMA tiles (4 x v_wmma_f32_16x16x32_bf16 per K-step).
// B K32xN64 tile staged in LDS transposed to [n][k]: each lane's B fragment
// is two 16-byte runs at k=8h.. and k=16+8h.. (two ds b128 reads). A fragments are two global
// b128 reads per lane when ACONT. K-tail (stem K=60) handled by a masked
// wave-uniform tail path; B OOB is zero-filled in LDS.
// Requires: M % 128 == 0 (true for every call here), N arbitrary mult. of 16.
// Implicit im2col: SAME conv of width kw => sAm = Cin, sAk = 1, K = kw*Cin
// on the zero-padded activation buffer.
// ---------------------------------------------------------------------------
template<bool ACONT>
__global__ __launch_bounds__(256) void k_gemm(
    const unsigned short* __restrict__ A, long long sAm, long long sAk,
    const unsigned short* __restrict__ B, long long sBk, long long sBn,
    const float* __restrict__ bias, float* __restrict__ C,
    int M, int N, int K, int ldc)
{
    __shared__ unsigned short Bs[64 * 32];    // [n][k], 4 KB

    const int tid  = threadIdx.x;
    const int lane = tid & 31;
    const int wid  = tid >> 5;
    const int wm   = wid & 3;
    const int wn   = wid >> 2;

    const int m0 = blockIdx.y * 128 + wm * 32;
    const int nb = blockIdx.x * 64;
    const int n0 = nb + wn * 32;

    const int half = lane >> 4;
    const int l15  = lane & 15;

    // cooperative B-fill mapping: thread -> (k row, 8 consecutive n)
    const int kf  = tid >> 3;          // 0..31
    const int nf0 = (tid & 7) * 8;     // 0..56

    v8f acc[2][2] = {};

    for (int kk = 0; kk < K; kk += 32) {
        const bool tail = (kk + 32 > K);   // wave-uniform (stem only)

        // ---- stage B tile into LDS, transposed ----
        {
            const int kg = kk + kf;
            if (!tail && sBn == 1 && (nb + nf0 + 7) < N) {
                const unsigned short* src = B + (long long)kg * sBk + (nb + nf0);
                uint4 v = *(const uint4*)src;
                unsigned short t8[8];
                *(uint4*)t8 = v;
#pragma unroll
                for (int i = 0; i < 8; ++i) Bs[(nf0 + i) * 32 + kf] = t8[i];
            } else {
#pragma unroll
                for (int i = 0; i < 8; ++i) {
                    int n = nb + nf0 + i;
                    unsigned short v = 0;
                    if (kg < K && n < N)
                        v = B[(long long)kg * sBk + (long long)n * sBn];
                    Bs[(nf0 + i) * 32 + kf] = v;
                }
            }
        }
        __syncthreads();

        // ---- A fragments (2 m-tiles per wave) ----
        v16bf afrag[2];
#pragma unroll
        for (int i = 0; i < 2; ++i) {
            const long long rowOff = (long long)(m0 + 16 * i + l15) * sAm;
            if (ACONT && !tail) {
                *(uint4*)&afrag[i]       = *(const uint4*)(A + rowOff + kk + 8 * half);
                *((uint4*)&afrag[i] + 1) = *(const uint4*)(A + rowOff + kk + 16 + 8 * half);
            } else if (!tail) {
#pragma unroll
                for (int j = 0; j < 8; ++j) {
                    int ka = kk + 16 * (j >> 2) + 2 * (j & 3) + 8 * half;
                    afrag[i][2 * j]     = us2bf(A[rowOff + (long long)ka       * sAk]);
                    afrag[i][2 * j + 1] = us2bf(A[rowOff + (long long)(ka + 1) * sAk]);
                }
            } else {
#pragma unroll
                for (int j = 0; j < 8; ++j) {
                    int ka = kk + 16 * (j >> 2) + 2 * (j & 3) + 8 * half;
                    unsigned short a0 = (ka     < K) ? A[rowOff + (long long)ka       * sAk] : (unsigned short)0;
                    unsigned short a1 = (ka + 1 < K) ? A[rowOff + (long long)(ka + 1) * sAk] : (unsigned short)0;
                    afrag[i][2 * j]     = us2bf(a0);
                    afrag[i][2 * j + 1] = us2bf(a1);
                }
            }
        }

        // ---- B fragments from LDS + 4 WMMAs ----
#pragma unroll
        for (int j = 0; j < 2; ++j) {
            const int nB = wn * 32 + 16 * j + l15;      // row in Bs
            v16bf bfrag;
            *(uint4*)&bfrag       = *(const uint4*)&Bs[nB * 32 + 8 * half];
            *((uint4*)&bfrag + 1) = *(const uint4*)&Bs[nB * 32 + 16 + 8 * half];
            acc[0][j] = __builtin_amdgcn_wmma_f32_16x16x32_bf16(
                            false, afrag[0], false, bfrag, (short)0, acc[0][j], false, false);
            acc[1][j] = __builtin_amdgcn_wmma_f32_16x16x32_bf16(
                            false, afrag[1], false, bfrag, (short)0, acc[1][j], false, false);
        }
        __syncthreads();
    }

    // ---- store (skip n-tiles beyond N; they accumulated on zero-filled B) --
#pragma unroll
    for (int j = 0; j < 2; ++j) {
        const int nn = n0 + 16 * j + l15;
        if (nn < N) {
            const float bv = bias ? bias[nn] : 0.f;
#pragma unroll
            for (int i = 0; i < 2; ++i) {
#pragma unroll
                for (int r = 0; r < 8; ++r) {
                    int m = m0 + 16 * i + r + 8 * half;
                    C[(long long)m * ldc + nn] = acc[i][j][r] + bv;
                }
            }
        }
    }
}

// ---------------------------------------------------------------------------
// elementwise / reduction kernels
// ---------------------------------------------------------------------------
__global__ void k_tobf(const float* __restrict__ s, unsigned short* __restrict__ d,
                       long long n, float scale)
{
    long long i = (long long)blockIdx.x * blockDim.x + threadIdx.x;
    if (i < n) d[i] = f2bf(s[i] * scale);
}

__global__ void k_zero_us(unsigned short* __restrict__ d, long long n)
{
    long long i = (long long)blockIdx.x * blockDim.x + threadIdx.x;
    if (i < n) d[i] = 0;
}

__global__ void k_add(float* __restrict__ x, const float* __restrict__ y, long long n)
{
    long long i = (long long)blockIdx.x * blockDim.x + threadIdx.x;
    if (i < n) x[i] += y[i];
}

__global__ void k_gelu32(const float* __restrict__ s, float* __restrict__ d, long long n)
{
    long long i = (long long)blockIdx.x * blockDim.x + threadIdx.x;
    if (i < n) d[i] = gelu_f(s[i]);
}

__global__ void k_gelu_bf(const float* __restrict__ s, unsigned short* __restrict__ d, long long n)
{
    long long i = (long long)blockIdx.x * blockDim.x + threadIdx.x;
    if (i < n) d[i] = f2bf(gelu_f(s[i]));
}

// per-channel mean/var over L (batchnorm axes (0,1)); one block per channel
__global__ void k_bn_stats(const float* __restrict__ x, int L, int C,
                           float* __restrict__ mean, float* __restrict__ var)
{
    int c = blockIdx.x;
    float s = 0.f, sq = 0.f;
    for (int l = threadIdx.x; l < L; l += blockDim.x) {
        float v = x[(long long)l * C + c];
        s += v; sq += v * v;
    }
    __shared__ float sh[256], sh2[256];
    sh[threadIdx.x] = s; sh2[threadIdx.x] = sq;
    __syncthreads();
    for (int o = blockDim.x / 2; o > 0; o >>= 1) {
        if ((int)threadIdx.x < o) { sh[threadIdx.x] += sh[threadIdx.x + o];
                                    sh2[threadIdx.x] += sh2[threadIdx.x + o]; }
        __syncthreads();
    }
    if (threadIdx.x == 0) {
        float m = sh[0] / (float)L;
        mean[c] = m;
        var[c]  = sh2[0] / (float)L - m * m;
    }
}

// bf16( gelu( batchnorm(x) ) ) written into padded GEMM input buffer
__global__ void k_bn_gelu(const float* __restrict__ x,
                          const float* __restrict__ mean, const float* __restrict__ var,
                          const float* __restrict__ g, const float* __restrict__ b,
                          unsigned short* __restrict__ out, long long LC, int C, int pad)
{
    long long i = (long long)blockIdx.x * blockDim.x + threadIdx.x;
    if (i >= LC) return;
    int c = (int)(i % C);
    float y = g[c] * (x[i] - mean[c]) * rsqrtf(var[c] + 1e-3f) + b[c];
    out[i + (long long)pad * C] = f2bf(gelu_f(y));
}

// softmax-pool (pool=2): combine positions 2*lo, 2*lo+1 with softmax(logits)
__global__ void k_pool(const float* __restrict__ x, const float* __restrict__ lg,
                       float* __restrict__ out, long long n, int C)
{
    long long i = (long long)blockIdx.x * blockDim.x + threadIdx.x;
    if (i >= n) return;
    long long lo = i / C; int c = (int)(i % C);
    long long i0 = (2 * lo) * C + c, i1 = i0 + C;
    float l0 = lg[i0], l1 = lg[i1];
    float m  = fmaxf(l0, l1);
    float e0 = expf(l0 - m), e1 = expf(l1 - m);
    out[i] = (x[i0] * e0 + x[i1] * e1) / (e0 + e1);
}

__global__ void k_addpe(float* __restrict__ x, int Nn, int D)
{
    int i = blockIdx.x * blockDim.x + threadIdx.x;
    if (i >= Nn * D) return;
    int n = i / D, d = i % D, half = D / 2;
    float pe;
    if (d < half) pe = sinf((float)n * powf(10000.f, -2.f * (float)d        / (float)D));
    else          pe = cosf((float)n * powf(10000.f, -2.f * (float)(d-half) / (float)D));
    x[i] += pe;
}

// layernorm over last dim; one block (D threads) per row
__global__ void k_layernorm(const float* __restrict__ x,
                            const float* __restrict__ g, const float* __restrict__ b,
                            float* __restrict__ out, int D)
{
    int n = blockIdx.x, d = threadIdx.x;
    float v = x[(long long)n * D + d];
    __shared__ float sh[128];
    sh[d] = v; __syncthreads();
    for (int o = D / 2; o > 0; o >>= 1) { if (d < o) sh[d] += sh[d + o]; __syncthreads(); }
    float m = sh[0] / (float)D;
    __syncthreads();
    float dv = v - m;
    sh[d] = dv * dv; __syncthreads();
    for (int o = D / 2; o > 0; o >>= 1) { if (d < o) sh[d] += sh[d + o]; __syncthreads(); }
    float var = sh[0] / (float)D;
    out[(long long)n * D + d] = g[d] * dv * rsqrtf(var + 1e-5f) + b[d];
}

// RoPE in place; x layout [n, h, dh]
__global__ void k_rope(float* __restrict__ x, int Nn, int H, int Dh)
{
    int i = blockIdx.x * blockDim.x + threadIdx.x;
    int hp = Dh / 2, tot = Nn * H * hp;
    if (i >= tot) return;
    int p = i % hp, rem = i / hp, h = rem % H, n = rem / H;
    long long base = (long long)n * (H * Dh) + h * Dh + 2 * p;
    float ang = (float)n * powf(10000.f, -(float)(2 * p) / (float)Dh);
    float c = cosf(ang), s = sinf(ang);
    float a = x[base], bb = x[base + 1];
    x[base]     = a * c - bb * s;
    x[base + 1] = a * s + bb * c;
}

// sum of squares per (h,n); x layout [n, h, dh]
__global__ void k_rowss(const float* __restrict__ x, float* __restrict__ ss,
                        int Nn, int H, int Dh)
{
    int i = blockIdx.x * blockDim.x + threadIdx.x;
    if (i >= H * Nn) return;
    int h = i / Nn, n = i % Nn;
    const float* p = x + (long long)n * (H * Dh) + h * Dh;
    float s = 0.f;
    for (int d = 0; d < Dh; ++d) s += p[d] * p[d];
    ss[i] = s;
}

__global__ void k_rowmax(const float* __restrict__ x, float* __restrict__ out,
                         int rows, int cols)
{
    int r = blockIdx.x * blockDim.x + threadIdx.x;
    if (r >= rows) return;
    const float* p = x + (long long)r * cols;
    float m = p[0];
    for (int c = 1; c < cols; ++c) m = fmaxf(m, p[c]);
    out[r] = m;
}

__global__ void k_headmax(const float* __restrict__ x, float* __restrict__ out, long long per)
{
    const float* p = x + (long long)blockIdx.x * per;
    float m = -3.4e38f;
    for (long long i = threadIdx.x; i < per; i += blockDim.x) m = fmaxf(m, p[i]);
    __shared__ float sh[256];
    sh[threadIdx.x] = m; __syncthreads();
    for (int o = blockDim.x / 2; o > 0; o >>= 1) {
        if ((int)threadIdx.x < o) sh[threadIdx.x] = fmaxf(sh[threadIdx.x], sh[threadIdx.x + o]);
        __syncthreads();
    }
    if (threadIdx.x == 0) out[blockIdx.x] = sh[0];
}

// Performer feature map: phi = ratio*(exp(xd - 0.5*dn^2*ss - stab) + STAB)
__global__ void k_phi(float* __restrict__ xd, const float* __restrict__ ss,
                      const float* __restrict__ stab, long long n,
                      int cols, int rowsPerHead, int perRowStab, float dn2h, float ratio)
{
    long long i = (long long)blockIdx.x * blockDim.x + threadIdx.x;
    if (i >= n) return;
    long long row = i / cols;
    float st = perRowStab ? stab[row] : stab[row / rowsPerHead];
    xd[i] = ratio * (expf(xd[i] - dn2h * ss[row] - st) + 0.001f);
}

// ksum[h,m] = sum_n kp[h][n,m]
__global__ void k_colsum(const float* __restrict__ kp, float* __restrict__ ksum,
                         int Nn, int Mf, int H)
{
    int i = blockIdx.x * blockDim.x + threadIdx.x;
    if (i >= H * Mf) return;
    int h = i / Mf, m = i % Mf;
    const float* p = kp + (long long)h * Nn * Mf + m;
    float s = 0.f;
    for (int n = 0; n < Nn; ++n) s += p[(long long)n * Mf];
    ksum[i] = s;
}

// den[h,n] = dot(qp[h,n,:], ksum[h,:]);  att[n, h*Dh + d] /= den
__global__ void k_denorm(const float* __restrict__ qp, const float* __restrict__ ksum,
                         float* __restrict__ att, int Nn, int Mf, int H, int Dh)
{
    int i = blockIdx.x * blockDim.x + threadIdx.x;
    if (i >= H * Nn) return;
    int h = i / Nn, n = i % Nn;
    const float* qr = qp + ((long long)h * Nn + n) * Mf;
    const float* ks = ksum + h * Mf;
    float den = 0.f;
    for (int m = 0; m < Mf; ++m) den += qr[m] * ks[m];
    float inv = 1.f / den;
    float* ar = att + (long long)n * (H * Dh) + h * Dh;
    for (int d = 0; d < Dh; ++d) ar[d] *= inv;
}

// ---------------------------------------------------------------------------
// host orchestration
// ---------------------------------------------------------------------------
extern "C" void kernel_launch(void* const* d_in, const int* in_sizes, int n_in,
                              void* d_out, int out_size, void* d_ws, size_t ws_size,
                              hipStream_t stream)
{
    (void)in_sizes; (void)n_in; (void)out_size; (void)ws_size;
    auto IN = [&](int i) -> const float* { return (const float*)d_in[i]; };

    // ---- workspace layout -------------------------------------------------
    const long long EMAX = 98304LL * 768;                 // largest L*C
    char* ws = (char*)d_ws;
    float*          Xb = (float*)ws;                      // activation ping
    float*          Yb = (float*)(ws + EMAX * 4);         // activation pong
    float*          Gb = (float*)(ws + 2 * EMAX * 4);     // pool logits
    unsigned short* P  = (unsigned short*)(ws + 3 * EMAX * 4);  // bf16 GEMM input
    size_t off = (size_t)(3 * EMAX * 4) + (size_t)(EMAX + 64) * 2;
    off = (off + 255) & ~(size_t)255;
    auto alloc = [&](size_t bytes) -> char* {
        char* p = ws + off; off = (off + bytes + 255) & ~(size_t)255; return p;
    };
    float*          meanb = (float*)alloc(1536 * 4);
    float*          varb  = (float*)alloc(1536 * 4);
    unsigned short* wbf   = (unsigned short*)alloc(9830400ULL * 2); // bf16 weights
    float*          h_    = (float*)alloc(98304 * 4);
    unsigned short* hb    = (unsigned short*)alloc(98304 * 2);
    float*          q_    = (float*)alloc(98304 * 4);
    float*          kk_   = (float*)alloc(98304 * 4);
    float*          v_    = (float*)alloc(98304 * 4);
    unsigned short* qb    = (unsigned short*)alloc(98304 * 2);
    unsigned short* kb    = (unsigned short*)alloc(98304 * 2);
    unsigned short* vb    = (unsigned short*)alloc(98304 * 2);
    float*          xdq   = (float*)alloc(786432 * 4);
    float*          xdk   = (float*)alloc(786432 * 4);
    unsigned short* qpb   = (unsigned short*)alloc(786432 * 2);
    unsigned short* kpb   = (unsigned short*)alloc(786432 * 2);
    float*          ssq   = (float*)alloc(3072 * 4);
    float*          ssk   = (float*)alloc(3072 * 4);
    float*          stq   = (float*)alloc(3072 * 4);
    float*          stk   = (float*)alloc(4 * 4);
    float*          ksum  = (float*)alloc(1024 * 4);
    float*          kv_   = (float*)alloc(32768 * 4);
    unsigned short* kvb   = (unsigned short*)alloc(32768 * 2);
    float*          att   = (float*)alloc(98304 * 4);
    unsigned short* attb  = (unsigned short*)alloc(98304 * 2);
    float*          tmp_  = (float*)alloc(98304 * 4);
    unsigned short* tmpb  = (unsigned short*)alloc(98304 * 2);
    unsigned short* pb    = (unsigned short*)alloc(8192 * 2);   // proj bf16

    auto cdiv = [](long long a, long long b) -> unsigned { return (unsigned)((a + b - 1) / b); };

    // dispatch: ACONT (sAk==1) picks the vectorized-A kernel
    auto gemm = [&](const unsigned short* A, long long sAm, long long sAk,
                    const unsigned short* B, long long sBk, long long sBn,
                    const float* bias, float* C, int M, int N, int K, int ldc) {
        dim3 grid(cdiv(N, 64), (unsigned)(M / 128));
        if (sAk == 1)
            k_gemm<true><<<grid, 256, 0, stream>>>(A, sAm, sAk, B, sBk, sBn, bias, C, M, N, K, ldc);
        else
            k_gemm<false><<<grid, 256, 0, stream>>>(A, sAm, sAk, B, sBk, sBn, bias, C, M, N, K, ldc);
    };
    auto tobf = [&](const float* s, unsigned short* d, long long n, float sc) {
        k_tobf<<<cdiv(n, 256), 256, 0, stream>>>(s, d, n, sc);
    };
    auto addv = [&](float* x, const float* y, long long n) {
        k_add<<<cdiv(n, 256), 256, 0, stream>>>(x, y, n);
    };

    // conv_block: y = conv1d(gelu(batchnorm(x)), wk, wb)
    auto conv_block = [&](const float* xin, int L, int Cin, int kw,
                          const float* bng, const float* bnb,
                          const float* wk, const float* wb, int Cout, float* out) {
        k_bn_stats<<<Cin, 256, 0, stream>>>(xin, L, Cin, meanb, varb);
        int pad = kw / 2;
        long long LC = (long long)L * Cin;
        if (pad) {
            long long pn = (long long)pad * Cin;
            k_zero_us<<<cdiv(pn, 256), 256, 0, stream>>>(P, pn);
            k_zero_us<<<cdiv(pn, 256), 256, 0, stream>>>(P + (long long)(L + pad) * Cin, pn);
        }
        k_bn_gelu<<<cdiv(LC, 256), 256, 0, stream>>>(xin, meanb, varb, bng, bnb, P, LC, Cin, pad);
        tobf(wk, wbf, (long long)kw * Cin * Cout, 1.f);
        gemm(P, Cin, 1, wbf, Cout, 1, wb, out, L, Cout, kw * Cin, Cout);
    };

    // softmax_pool (pool=2) with general C x C logit weight
    auto pool = [&](const float* xin, int L, int C, const float* w, float* out) {
        tobf(xin, P, (long long)L * C, 1.f);
        tobf(w, wbf, (long long)C * C, 1.f);
        gemm(P, C, 1, wbf, C, 1, nullptr, Gb, L, C, C, C);
        long long n = (long long)(L / 2) * C;
        k_pool<<<cdiv(n, 256), 256, 0, stream>>>(xin, Gb, out, n, C);
    };

    float* X = Xb;
    float* Y = Yb;

    // ---- stem conv: 15-wide, 4 -> 768 ------------------------------------
    {
        const int L = 98304, Cin = 4, Cout = 768, kw = 15, pad = 7;
        long long pn = (long long)pad * Cin;
        k_zero_us<<<cdiv(pn, 256), 256, 0, stream>>>(P, pn);
        k_zero_us<<<cdiv(pn, 256), 256, 0, stream>>>(P + (long long)(L + pad) * Cin, pn);
        tobf(IN(0), P + pn, (long long)L * Cin, 1.f);            // sequence
        tobf(IN(1), wbf, (long long)kw * Cin * Cout, 1.f);       // stem_k
        gemm(P, Cin, 1, wbf, Cout, 1, IN(2), X, L, Cout, kw * Cin, Cout);
    }
    // stem residual block + pool
    conv_block(X, 98304, 768, 1, IN(3), IN(4), IN(5), IN(6), 768, Y);
    addv(X, Y, 98304LL * 768);
    pool(X, 98304, 768, IN(7), Y);
    { float* t = X; X = Y; Y = t; }

    // ---- tower ------------------------------------------------------------
    const int FL[6] = {768, 896, 1024, 1152, 1280, 1536};
    int cin = 768, L = 49152;
    for (int s = 0; s < 6; ++s) {
        int f = FL[s], b = 8 + s * 9;
        conv_block(X, L, cin, 5, IN(b + 0), IN(b + 1), IN(b + 2), IN(b + 3), f, Y);
        { float* t = X; X = Y; Y = t; }
        conv_block(X, L, f, 1, IN(b + 4), IN(b + 5), IN(b + 6), IN(b + 7), f, Y);
        addv(X, Y, (long long)L * f);
        pool(X, L, f, IN(b + 8), Y);
        { float* t = X; X = Y; Y = t; }
        L /= 2; cin = f;
    }
    // L = 768, C = 1536

    // ---- mix 1536 -> 128 + sinusoidal PE ---------------------------------
    tobf(X, P, (long long)L * 1536, 1.f);
    tobf(IN(62), wbf, 1536LL * 128, 1.f);
    gemm(P, 1536, 1, wbf, 128, 1, IN(63), Y, L, 128, 1536, 128);
    { float* t = X; X = Y; Y = t; }
    k_addpe<<<cdiv(768LL * 128, 256), 256, 0, stream>>>(X, 768, 128);

    // ---- transformer (Performer/FAVOR+), 8 layers ------------------------
    const int Nn = 768, D = 128, H = 4, Dh = 32, Mf = 256;
    const float dn = 1.f / powf((float)Dh, 0.25f);
    const float dn2h = 0.5f * dn * dn;
    const float ratio = 1.f / sqrtf((float)Mf);
    tobf(IN(82), pb, (long long)Mf * Dh, 1.f);                   // proj

    for (int l = 0; l < 8; ++l) {
        const float* ln1g = IN(68) + l * D;  const float* ln1b = IN(69) + l * D;
        const float* wq = IN(70) + l * D * D; const float* wk = IN(71) + l * D * D;
        const float* wv = IN(72) + l * D * D; const float* wo = IN(73) + l * D * D;
        const float* ln2g = IN(74) + l * D;  const float* ln2b = IN(75) + l * D;
        const float* w1 = IN(76) + l * D * D; const float* b1 = IN(77) + l * D;
        const float* w2 = IN(78) + l * D * D; const float* b2 = IN(79) + l * D;

        k_layernorm<<<Nn, D, 0, stream>>>(X, ln1g, ln1b, h_, D);
        tobf(h_, hb, (long long)Nn * D, 1.f);
        tobf(wq, wbf, D * D, 1.f);
        gemm(hb, D, 1, wbf, D, 1, nullptr, q_, Nn, D, D, D);
        tobf(wk, wbf, D * D, 1.f);
        gemm(hb, D, 1, wbf, D, 1, nullptr, kk_, Nn, D, D, D);
        tobf(wv, wbf, D * D, 1.f);
        gemm(hb, D, 1, wbf, D, 1, nullptr, v_, Nn, D, D, D);

        k_rope<<<cdiv((long long)Nn * H * (Dh / 2), 256), 256, 0, stream>>>(q_, Nn, H, Dh);
        k_rope<<<cdiv((long long)Nn * H * (Dh / 2), 256), 256, 0, stream>>>(kk_, Nn, H, Dh);
        k_rowss<<<cdiv(H * Nn, 256), 256, 0, stream>>>(q_, ssq, Nn, H, Dh);
        k_rowss<<<cdiv(H * Nn, 256), 256, 0, stream>>>(kk_, ssk, Nn, H, Dh);
        tobf(q_, qb, (long long)Nn * D, dn);     // fold x*dn into bf16 pack
        tobf(kk_, kb, (long long)Nn * D, dn);
        tobf(v_, vb, (long long)Nn * D, 1.f);

        for (int h = 0; h < H; ++h) {
            // xd[h] = (x*dn) @ proj^T :  B(k,n) = proj[n*Dh + k]
            gemm(qb + h * Dh, D, 1, pb, 1, Dh, nullptr, xdq + (long long)h * Nn * Mf, Nn, Mf, Dh, Mf);
            gemm(kb + h * Dh, D, 1, pb, 1, Dh, nullptr, xdk + (long long)h * Nn * Mf, Nn, Mf, Dh, Mf);
        }
        k_rowmax<<<cdiv(H * Nn, 64), 64, 0, stream>>>(xdq, stq, H * Nn, Mf);
        k_headmax<<<H, 256, 0, stream>>>(xdk, stk, (long long)Nn * Mf);
        k_phi<<<cdiv((long long)H * Nn * Mf, 256), 256, 0, stream>>>(
            xdq, ssq, stq, (long long)H * Nn * Mf, Mf, Nn, 1, dn2h, ratio);
        k_phi<<<cdiv((long long)H * Nn * Mf, 256), 256, 0, stream>>>(
            xdk, ssk, stk, (long long)H * Nn * Mf, Mf, Nn, 0, dn2h, ratio);
        tobf(xdq, qpb, (long long)H * Nn * Mf, 1.f);
        tobf(xdk, kpb, (long long)H * Nn * Mf, 1.f);

        k_colsum<<<cdiv(H * Mf, 64), 64, 0, stream>>>(xdk, ksum, Nn, Mf, H);
        for (int h = 0; h < H; ++h) {
            // kv[h] = kp[h]^T @ v[h] :  A(m,k) = kp[k*Mf + m]  (strided transpose)
            gemm(kpb + (long long)h * Nn * Mf, 1, Mf, vb + h * Dh, D, 1, nullptr,
                 kv_ + h * Mf * Dh, Mf, Dh, Nn, Dh);
        }
        tobf(kv_, kvb, H * Mf * Dh, 1.f);
        for (int h = 0; h < H; ++h) {
            // num[h] = qp[h] @ kv[h], written interleaved into att[n, h*Dh + d]
            gemm(qpb + (long long)h * Nn * Mf, Mf, 1, kvb + h * Mf * Dh, Dh, 1, nullptr,
                 att + h * Dh, Nn, Dh, Mf, D);
        }
        k_denorm<<<cdiv(H * Nn, 64), 64, 0, stream>>>(xdq, ksum, att, Nn, Mf, H, Dh);

        tobf(att, attb, (long long)Nn * D, 1.f);
        tobf(wo, wbf, D * D, 1.f);
        gemm(attb, D, 1, wbf, D, 1, nullptr, tmp_, Nn, D, D, D);
        addv(X, tmp_, (long long)Nn * D);

        // FFN
        k_layernorm<<<Nn, D, 0, stream>>>(X, ln2g, ln2b, h_, D);
        tobf(h_, hb, (long long)Nn * D, 1.f);
        tobf(w1, wbf, D * D, 1.f);
        gemm(hb, D, 1, wbf, D, 1, b1, tmp_, Nn, D, D, D);
        k_gelu_bf<<<cdiv((long long)Nn * D, 256), 256, 0, stream>>>(tmp_, tmpb, (long long)Nn * D);
        tobf(w2, wbf, D * D, 1.f);
        gemm(tmpb, D, 1, wbf, D, 1, b2, tmp_, Nn, D, D, D);
        addv(X, tmp_, (long long)Nn * D);
    }

    // ---- final: layernorm -> conv_block -> gelu --------------------------
    k_layernorm<<<Nn, D, 0, stream>>>(X, IN(80), IN(81), h_, D);
    conv_block(h_, Nn, D, 1, IN(64), IN(65), IN(66), IN(67), 16, tmp_);
    k_gelu32<<<cdiv((long long)Nn * 16, 256), 256, 0, stream>>>(tmp_, (float*)d_out, (long long)Nn * 16);
}
